// CFConv_13245679141058
// MI455X (gfx1250) — matrix operand order repaired and checked by reading, not compiled, not run on hardware
//
#include <hip/hip_runtime.h>
#include <hip/hip_bf16.h>

// ---------------------------------------------------------------------------
// MI455X (gfx1250) fused SchNet CFConv.
// Precision choice: bf16 WMMA (v_wmma_f32_16x16x32_bf16) with f32 accum.
//   FLOP budget ~20 GFLOP dominated by the per-pair 128x128 filter layer;
//   HBM floor ~64MB => the matrix path, not bandwidth, is the limiter, so we
//   run everything through the 16x16x32 bf16 WMMA pipe.
// Data movement: weights pre-swizzled once into B-fragment layout (bf16),
//   filter activations transposed between layer1(C-layout) and layer2
//   (A-layout) through padded LDS; neighbor-feature gather is L2-resident.
// Activation: shifted softplus via hardware v_exp_f32/v_log_f32 (TRANS ops
//   co-issue with WMMA), stable form max(v,0)+ln(1+e^-|v|)-ln2 -- avoids the
//   ~50-instruction precise log1pf expansion seen in the previous round.
// ---------------------------------------------------------------------------

typedef __attribute__((ext_vector_type(16))) __bf16 v16bf;
typedef __attribute__((ext_vector_type(8)))  __bf16 v8bf;
typedef __attribute__((ext_vector_type(8)))  float  v8f;

#define N_ATOMS   10000
#define N_NBH     48
#define N_F       128
#define N_GAUSS   25
#define LOG2_F    0.6931471805599453f

__device__ __forceinline__ v8f wmma_bf16(v16bf a, v16bf b, v8f c) {
    // 8 args: (neg_a, A, neg_b, B, c_mod, C, reuse_a, reuse_b)
    return __builtin_amdgcn_wmma_f32_16x16x32_bf16(false, a, false, b,
                                                   (short)0, c, false, false);
}

__device__ __forceinline__ float ssp(float v) {
    // shifted softplus: ln(1+e^v) - ln2
    // stable: max(v,0) + ln(1 + e^-|v|) - ln2 ; e^-|v| in (0,1] so the log
    // argument is in (1,2] -> safe for the fast hardware log.
    float t = __expf(-fabsf(v));        // v_exp_f32
    return fmaxf(v, 0.0f) + __logf(1.0f + t) - LOG2_F;   // v_log_f32
}

// Build an A fragment (16x32 bf16, lane-striped per ISA layout) from a
// row-major f32 row pointer.  lane<16 holds K {kg+0..7, kg+16..23} of row
// lane; lanes>=16 hold K {kg+8..15, kg+24..31} of row lane-16.
__device__ __forceinline__ v16bf load_a_rowmajor_f32(const float* rowptr,
                                                     int lane, int kt) {
    const int hi  = (lane & 16) ? 1 : 0;
    const int kg0 = kt * 32 + (hi ? 8 : 0);
    const int kg1 = kt * 32 + (hi ? 24 : 16);
    const float4* p0 = (const float4*)(rowptr + kg0);
    const float4* p1 = (const float4*)(rowptr + kg1);
    float4 x0 = p0[0], x1 = p0[1];
    float4 y0 = p1[0], y1 = p1[1];
    v16bf a;
    a[0]  = (__bf16)x0.x; a[1]  = (__bf16)x0.y; a[2]  = (__bf16)x0.z; a[3]  = (__bf16)x0.w;
    a[4]  = (__bf16)x1.x; a[5]  = (__bf16)x1.y; a[6]  = (__bf16)x1.z; a[7]  = (__bf16)x1.w;
    a[8]  = (__bf16)y0.x; a[9]  = (__bf16)y0.y; a[10] = (__bf16)y0.z; a[11] = (__bf16)y0.w;
    a[12] = (__bf16)y1.x; a[13] = (__bf16)y1.y; a[14] = (__bf16)y1.z; a[15] = (__bf16)y1.w;
    return a;
}

// ---------------------------------------------------------------------------
// Kernel 0: swizzle the 4 weight matrices (f32 row-major [K,128]) into bf16
// WMMA B-fragment layout: per 32x16 tile, lane l stores its 16 contiguous
// bf16 values (col = nt*16 + l%16, K = kt*32 + (l<16?0:8) + (e&7) + (e>=8?16:0)).
// Blocks: 0-31 W_in2f, 32-39 W1 (K=25, zero-padded), 40-71 W2, 72-103 W_f2out.
// ---------------------------------------------------------------------------
__global__ __launch_bounds__(512) void swizzle_weights_kernel(
    const float* W_in2f, const float* W1, const float* W2, const float* W_f2out,
    __bf16* fIn2f, __bf16* fW1, __bf16* fW2, __bf16* fOut) {
    int b = blockIdx.x;
    const float* src; __bf16* dst; int Ks; int t;
    if (b < 32)      { src = W_in2f;  dst = fIn2f; Ks = 128; t = b;      }
    else if (b < 40) { src = W1;      dst = fW1;   Ks = 25;  t = b - 32; }
    else if (b < 72) { src = W2;      dst = fW2;   Ks = 128; t = b - 40; }
    else             { src = W_f2out; dst = fOut;  Ks = 128; t = b - 72; }
    int l  = threadIdx.x >> 4;     // lane 0..31
    int e  = threadIdx.x & 15;     // element 0..15
    int kt = t >> 3, nt = t & 7;
    int k   = kt * 32 + ((l & 16) ? 8 : 0) + (e & 7) + ((e & 8) ? 16 : 0);
    int col = nt * 16 + (l & 15);
    float v = (k < Ks) ? src[(size_t)k * N_F + col] : 0.0f;
    dst[(size_t)t * 512 + l * 16 + e] = (__bf16)v;
}

// ---------------------------------------------------------------------------
// Kernels 1 & 3: 128-wide GEMM  out[M,128] = A[M,128] @ Bfrag, optional
// bias + shifted softplus.  One 16-row M-tile per wave, 4 waves per block.
// ---------------------------------------------------------------------------
template <bool BIAS_SSP>
__global__ __launch_bounds__(128) void gemm128_kernel(
    const float* __restrict__ A, const __bf16* __restrict__ Bfrag,
    const float* __restrict__ bias, float* __restrict__ out, int Mtiles) {
    const int wave = threadIdx.x >> 5;
    const int lane = threadIdx.x & 31;
    const int mt   = blockIdx.x * 4 + wave;
    if (mt >= Mtiles) return;                 // wave-uniform exit

    const int col0 = lane & 15;
    const int row  = mt * 16 + col0;          // A-fragment source row
    const float* rowptr = A + (size_t)row * N_F;

    v16bf afr[4];
    #pragma unroll
    for (int kt = 0; kt < 4; ++kt) afr[kt] = load_a_rowmajor_f32(rowptr, lane, kt);

    const int rbase = mt * 16 + ((lane & 16) ? 8 : 0);
    #pragma unroll
    for (int nt = 0; nt < 8; ++nt) {
        v8f c = {};
        #pragma unroll
        for (int kt = 0; kt < 4; ++kt) {
            const v16bf* tile = (const v16bf*)(Bfrag + (size_t)(kt * 8 + nt) * 512);
            c = wmma_bf16(afr[kt], tile[lane], c);
        }
        const int col = nt * 16 + col0;
        float bv = BIAS_SSP ? bias[col] : 0.0f;
        #pragma unroll
        for (int j = 0; j < 8; ++j) {
            float v = c[j] + bv;
            if (BIAS_SSP) v = ssp(v);
            out[(size_t)(rbase + j) * N_F + col] = v;
        }
    }
}

// ---------------------------------------------------------------------------
// Kernel 2: fused filter network + cutoff + gather + neighbor reduction.
// One wave per atom, 4 atoms per 128-thread workgroup, 2500 workgroups.
// ---------------------------------------------------------------------------
__global__ __launch_bounds__(128) void cfconv_fused_kernel(
    const float* __restrict__ dRexp, const float* __restrict__ dR,
    const float* __restrict__ pmask, const int* __restrict__ nbrIdx,
    const __bf16* __restrict__ w1frag, const float* __restrict__ b1,
    const __bf16* __restrict__ w2frag, const float* __restrict__ b2,
    const float* __restrict__ y_ws, float* __restrict__ agg) {
    // 272-byte row stride (128+8 bf16) => consecutive rows land on different
    // bank groups for the per-lane b128 A-fragment reads.
    __shared__ __align__(16) __bf16 actS[4][N_NBH][N_F + 8];
    __shared__ float cmS[4][N_NBH];
    __shared__ int   nbS[4][N_NBH];

    const int wave = threadIdx.x >> 5;
    const int lane = threadIdx.x & 31;
    const int atom = blockIdx.x * 4 + wave;        // < 10000 by construction
    const int col0 = lane & 15;
    const int hi   = (lane & 16) ? 1 : 0;

    // --- neighbor metadata: cutoff * pairwise mask, gather indices ---------
    for (int k = lane; k < N_NBH; k += 32) {
        float d  = dR[(size_t)atom * N_NBH + k];
        float cm = (d <= 5.0f ? 1.0f : 0.0f) * pmask[(size_t)atom * N_NBH + k];
        cmS[wave][k] = cm;
        nbS[wave][k] = nbrIdx[(size_t)atom * N_NBH + k];
    }

    float b2v[8];
    #pragma unroll
    for (int nt = 0; nt < 8; ++nt) b2v[nt] = b2[nt * 16 + col0];

    // --- layer 1: [48,25(pad32)] @ W1 -> ssp -> bf16 in LDS ---------------
    #pragma unroll
    for (int mt = 0; mt < 3; ++mt) {
        const int row = mt * 16 + col0;
        const float* rp = dRexp + ((size_t)atom * N_NBH + row) * N_GAUSS;
        const int kg0 = hi ? 8 : 0, kg1 = hi ? 24 : 16;
        v16bf a1;
        #pragma unroll
        for (int e = 0; e < 8; ++e) {
            int k0 = kg0 + e, k1 = kg1 + e;
            a1[e]     = (__bf16)(k0 < N_GAUSS ? rp[k0] : 0.0f);
            a1[8 + e] = (__bf16)(k1 < N_GAUSS ? rp[k1] : 0.0f);
        }
        const int r0 = mt * 16 + (hi ? 8 : 0);
        #pragma unroll
        for (int nt = 0; nt < 8; ++nt) {
            const v16bf* tile = (const v16bf*)(w1frag + (size_t)nt * 512);
            v8f c = {};
            c = wmma_bf16(a1, tile[lane], c);
            float bv = b1[nt * 16 + col0];
            #pragma unroll
            for (int j = 0; j < 8; ++j)
                actS[wave][r0 + j][nt * 16 + col0] = (__bf16)ssp(c[j] + bv);
        }
    }

    // --- layer 2 + cutoff + gather + reduce over 48 neighbors -------------
    float accF[8];
    #pragma unroll
    for (int i = 0; i < 8; ++i) accF[i] = 0.0f;

    #pragma unroll
    for (int mt = 0; mt < 3; ++mt) {
        const int row = mt * 16 + col0;
        v16bf a2[4];
        #pragma unroll
        for (int kt = 0; kt < 4; ++kt) {
            const int kg0 = kt * 32 + (hi ? 8 : 0);
            const int kg1 = kt * 32 + (hi ? 24 : 16);
            v8bf lo = *(const v8bf*)&actS[wave][row][kg0];
            v8bf hv = *(const v8bf*)&actS[wave][row][kg1];
            #pragma unroll
            for (int e = 0; e < 8; ++e) { a2[kt][e] = lo[e]; a2[kt][8 + e] = hv[e]; }
        }
        const int r0 = mt * 16 + (hi ? 8 : 0);
        #pragma unroll
        for (int nt = 0; nt < 8; ++nt) {
            v8f c = {};
            #pragma unroll
            for (int kt = 0; kt < 4; ++kt) {
                const v16bf* tile = (const v16bf*)(w2frag + (size_t)(kt * 8 + nt) * 512);
                c = wmma_bf16(a2[kt], tile[lane], c);
            }
            const int col = nt * 16 + col0;
            float s = 0.0f;
            #pragma unroll
            for (int j = 0; j < 8; ++j) {
                const int k = r0 + j;                      // neighbor slot
                float w  = (c[j] + b2v[nt]) * cmS[wave][k];
                float yv = y_ws[(size_t)nbS[wave][k] * N_F + col];
                s = fmaf(w, yv, s);
            }
            accF[nt] += s;
        }
    }

    // columns are duplicated in lane l and l+16 with disjoint row halves:
    // add across the two half-waves, lanes 0..15 write the result.
    #pragma unroll
    for (int nt = 0; nt < 8; ++nt) {
        float v = accF[nt] + __shfl_xor(accF[nt], 16, 32);
        if (lane < 16) agg[(size_t)atom * N_F + nt * 16 + lane] = v;
    }
}

// ---------------------------------------------------------------------------
// Workspace layout (bytes):
//   [0, 5.12MB)        y_ws   : x @ W_in2f, f32 [10000,128]
//   [5.12MB, 10.24MB)  agg    : aggregated features, f32 [10000,128]
//   [10.24MB, +32KB)   fIn2f  : W_in2f bf16 fragments (32 tiles)
//   [+8KB)             fW1    : W1 bf16 fragments (8 tiles, K padded 25->32)
//   [+32KB)            fW2
//   [+32KB)            fOut
// Total ~10.35 MB.
// ---------------------------------------------------------------------------
extern "C" void kernel_launch(void* const* d_in, const int* in_sizes, int n_in,
                              void* d_out, int out_size, void* d_ws, size_t ws_size,
                              hipStream_t stream) {
    const float* x       = (const float*)d_in[0];
    const float* dR      = (const float*)d_in[1];
    const float* dRexp   = (const float*)d_in[2];
    const float* pmask   = (const float*)d_in[3];
    const int*   nbr     = (const int*)d_in[4];
    const float* W1      = (const float*)d_in[5];
    const float* b1      = (const float*)d_in[6];
    const float* W2      = (const float*)d_in[7];
    const float* b2      = (const float*)d_in[8];
    const float* W_in2f  = (const float*)d_in[9];
    const float* W_f2out = (const float*)d_in[10];
    const float* b_f2out = (const float*)d_in[11];
    float* out = (float*)d_out;

    char* ws = (char*)d_ws;
    float*  y_ws  = (float*)(ws);
    float*  agg   = (float*)(ws + 5120000);
    __bf16* fIn2f = (__bf16*)(ws + 10240000);
    __bf16* fW1   = (__bf16*)(ws + 10272768);
    __bf16* fW2   = (__bf16*)(ws + 10280960);
    __bf16* fOut  = (__bf16*)(ws + 10313728);

    (void)in_sizes; (void)n_in; (void)out_size; (void)ws_size;

    // 0) weights -> bf16 WMMA B-fragment layout (L2-resident thereafter)
    swizzle_weights_kernel<<<104, 512, 0, stream>>>(W_in2f, W1, W2, W_f2out,
                                                    fIn2f, fW1, fW2, fOut);
    // 1) y = x @ W_in2f                (625 M-tiles, 4 waves/block)
    gemm128_kernel<false><<<157, 128, 0, stream>>>(x, fIn2f, nullptr, y_ws, 625);
    // 2) fused filter net + cutoff + gather + neighbor sum (1 wave / atom)
    cfconv_fused_kernel<<<2500, 128, 0, stream>>>(dRexp, dR, pmask, nbr,
                                                  fW1, b1, fW2, b2, y_ws, agg);
    // 3) out = ssp(agg @ W_f2out + b)
    gemm128_kernel<true><<<157, 128, 0, stream>>>(agg, fOut, b_f2out, out, 625);
}
